// Tracker_1786706395211
// MI455X (gfx1250) — matrix operand order repaired and checked by reading, not compiled, not run on hardware
//
#include <hip/hip_runtime.h>
#include <math.h>

typedef __attribute__((ext_vector_type(16))) _Float16 v16h;
typedef __attribute__((ext_vector_type(8)))  float    v8f;
typedef _Float16 half_t;

#define T_FRAMES 16
#define NPTS     128
#define LATC     128
#define RR       7
#define SUP      49
#define KP_CV    2432   // 2401 padded to 32
#define KP_X     1120   // 1110 padded to 32
#define D_INF    1110
#define H0       96
#define W0       128

union FragH { v16h v; uint4 q[2]; };

static __device__ __forceinline__ float gelu_exact(float x) {
    return 0.5f * x * (1.0f + erff(x * 0.70710678118654752f));
}

// ---------------------------------------------------------------- weights->f16
__global__ void cvt_pad_f16(const float* __restrict__ src, half_t* __restrict__ dst,
                            int srcRows, int dstRows, int cols) {
    int idx = blockIdx.x * blockDim.x + threadIdx.x;
    if (idx >= dstRows * cols) return;
    int r = idx / cols, c = idx % cols;
    dst[idx] = (r < srcRows) ? (half_t)src[r * cols + c] : (half_t)0.f;
}

// ---------------------------------------------------------------- time emb interp (50 -> 16)
__global__ void time_interp(const float* __restrict__ emb, float* __restrict__ te) {
    int idx = blockIdx.x * blockDim.x + threadIdx.x;
    if (idx >= T_FRAMES * D_INF) return;
    int t = idx / D_INF, c = idx % D_INF;
    float sc = 50.f / 16.f;
    float j = fminf(fmaxf((t + 0.5f) * sc - 0.5f, 0.f), 49.f);
    int j0 = (int)floorf(j);
    int j1 = min(j0 + 1, 49);
    float w = j - (float)j0;
    te[idx] = emb[j0 * D_INF + c] * (1.f - w) + emb[j1 * D_INF + c] * w;
}

// ---------------------------------------------------------------- fnet conv + L2 norm
__global__ __launch_bounds__(128)
void fnet_conv(const float* __restrict__ video, const float* __restrict__ w,
               const float* __restrict__ b, half_t* __restrict__ pyr0) {
    __shared__ float patch[48];
    __shared__ float red[128];
    int bid = blockIdx.x;                    // t*H0*W0 + y*W0 + x
    int x = bid & (W0 - 1);
    int y = (bid >> 7) % H0;
    int t = bid / (H0 * W0);
    int ch = threadIdx.x;
    if (ch < 48) {
        int c  = ch >> 4;
        int ky = (ch >> 2) & 3;
        int kx = ch & 3;
        float v = video[(((size_t)t * 3 + c) * 384 + (4 * y + ky)) * 512 + (4 * x + kx)];
        patch[ch] = 2.f * (v / 255.f) - 1.f;
    }
    __syncthreads();
    float acc = b[ch];
    const float* wc = w + ch * 48;
#pragma unroll
    for (int p = 0; p < 48; ++p) acc += patch[p] * wc[p];
    red[ch] = acc * acc;
    __syncthreads();
    for (int st = 64; st > 0; st >>= 1) {
        if (ch < st) red[ch] += red[ch + st];
        __syncthreads();
    }
    float nrm = rsqrtf(fmaxf(red[0], 1e-12f));
    pyr0[(size_t)bid * LATC + ch] = (half_t)(acc * nrm);
}

// ---------------------------------------------------------------- 2x2 avg pool
__global__ __launch_bounds__(128)
void pool2(const half_t* __restrict__ in, half_t* __restrict__ out, int Hi, int Wi) {
    int Ho = Hi >> 1, Wo = Wi >> 1;
    int bid = blockIdx.x;                    // t*Ho*Wo + y*Wo + x
    int x = bid % Wo;
    int y = (bid / Wo) % Ho;
    int t = bid / (Wo * Ho);
    int ch = threadIdx.x;
    const half_t* base = in + (size_t)t * Hi * Wi * LATC;
    float s = (float)base[((size_t)(2 * y) * Wi + 2 * x) * LATC + ch]
            + (float)base[((size_t)(2 * y) * Wi + 2 * x + 1) * LATC + ch]
            + (float)base[((size_t)(2 * y + 1) * Wi + 2 * x) * LATC + ch]
            + (float)base[((size_t)(2 * y + 1) * Wi + 2 * x + 1) * LATC + ch];
    out[(size_t)bid * LATC + ch] = (half_t)(0.25f * s);
}

__device__ __forceinline__ void lvl_info(int lvl, int& Hl, int& Wl, size_t& off) {
    const int hs[4] = {96, 48, 24, 12};
    const int wz[4] = {128, 64, 32, 16};
    Hl = hs[lvl]; Wl = wz[lvl];
    size_t o = 0;
    for (int i = 0; i < lvl; ++i) o += (size_t)T_FRAMES * hs[i] * wz[i] * LATC;
    off = o;
}

// ---------------------------------------------------------------- template features (trilinear t,y,x)
__global__ __launch_bounds__(128)
void sample_tfs(const half_t* __restrict__ pyr, const float* __restrict__ queries,
                half_t* __restrict__ tfs) {
    int bid = blockIdx.x;                    // lvl*N*49 + n*49 + s
    int s = bid % SUP;
    int rem = bid / SUP;
    int n = rem % NPTS;
    int lvl = rem / NPTS;
    int ch = threadIdx.x;

    int Hl, Wl; size_t off;
    lvl_info(lvl, Hl, Wl, off);
    float inv = 1.f / (float)(1 << lvl);

    float tf = queries[n * 3 + 0];
    float cx = queries[n * 3 + 1] * 0.25f * inv + (float)(s / RR - 3);
    float cy = queries[n * 3 + 2] * 0.25f * inv + (float)(s % RR - 3);
    tf = fminf(fmaxf(tf, 0.f), (float)(T_FRAMES - 1));
    cx = fminf(fmaxf(cx, 0.f), (float)(Wl - 1));
    cy = fminf(fmaxf(cy, 0.f), (float)(Hl - 1));

    int t0 = (int)floorf(tf); int t1 = min(t0 + 1, T_FRAMES - 1); float wt = tf - t0;
    int x0 = (int)floorf(cx); int x1 = min(x0 + 1, Wl - 1);       float wx = cx - x0;
    int y0 = (int)floorf(cy); int y1 = min(y0 + 1, Hl - 1);       float wy = cy - y0;

    const half_t* base = pyr + off;
    auto G = [&](int tt, int yy, int xx) {
        return (float)base[(((size_t)tt * Hl + yy) * Wl + xx) * LATC + ch];
    };
    float b0 = G(t0, y0, x0) * (1 - wy) * (1 - wx) + G(t0, y0, x1) * (1 - wy) * wx
             + G(t0, y1, x0) * wy * (1 - wx)       + G(t0, y1, x1) * wy * wx;
    float b1 = G(t1, y0, x0) * (1 - wy) * (1 - wx) + G(t1, y0, x1) * (1 - wy) * wx
             + G(t1, y1, x0) * wy * (1 - wx)       + G(t1, y1, x1) * wy * wx;
    tfs[(size_t)bid * LATC + ch] = (half_t)(b0 * (1 - wt) + b1 * wt);
}

// ---------------------------------------------------------------- current features (bilinear within frame)
__global__ __launch_bounds__(128)
void sample_cf(const half_t* __restrict__ pyrLvl, const float* __restrict__ coords,
               half_t* __restrict__ cf, int lvl, int Hl, int Wl) {
    int bid = blockIdx.x;                    // (t*N+n)*49 + s
    int s = bid % SUP;
    int tn = bid / SUP;
    int t = tn >> 7;
    int ch = threadIdx.x;

    float inv = 1.f / (float)(1 << lvl);
    float cx = coords[tn * 2 + 0] * inv + (float)(s / RR - 3);
    float cy = coords[tn * 2 + 1] * inv + (float)(s % RR - 3);
    cx = fminf(fmaxf(cx, 0.f), (float)(Wl - 1));
    cy = fminf(fmaxf(cy, 0.f), (float)(Hl - 1));
    int x0 = (int)floorf(cx); int x1 = min(x0 + 1, Wl - 1); float wx = cx - x0;
    int y0 = (int)floorf(cy); int y1 = min(y0 + 1, Hl - 1); float wy = cy - y0;

    const half_t* base = pyrLvl + (size_t)t * Hl * Wl * LATC;
    auto G = [&](int yy, int xx) {
        return (float)base[((size_t)yy * Wl + xx) * LATC + ch];
    };
    float v = G(y0, x0) * (1 - wy) * (1 - wx) + G(y0, x1) * (1 - wy) * wx
            + G(y1, x0) * wy * (1 - wx)       + G(y1, x1) * wy * wx;
    cf[((size_t)tn * SUP + s) * LATC + ch] = (half_t)v;
}

// ---------------------------------------------------------------- correlation: cv[49x49] = cf[49x128] @ tfs^T, WMMA
// A (cf) staged M x K row-major; B^T (= tfs rows, N x K) staged row-major.
// All fragment reads are contiguous 16B chunks -> ds_load_b128.
__global__ __launch_bounds__(256)
void cv_wmma(const half_t* __restrict__ cf, const half_t* __restrict__ tfs,
             half_t* __restrict__ CV, int lvl) {
    __shared__ __align__(16) half_t As[64][136];   // cf rows, rows 49..63 zero (stride 272B)
    __shared__ __align__(16) half_t Bt[64][136];   // tfs rows (= B^T), rows 49..63 zero
    int bid = blockIdx.x;                    // t*N+n
    int n = bid & (NPTS - 1);
    int tid = threadIdx.x;
    int lane = tid & 31, wave = tid >> 5;

    const half_t* cfb = cf + (size_t)bid * SUP * LATC;
    const half_t* tb  = tfs + (size_t)(lvl * NPTS + n) * SUP * LATC;
    const uint4 zz = make_uint4(0u, 0u, 0u, 0u);
#pragma unroll
    for (int i = 0; i < 4; ++i) {            // 1024 chunks of 8 halves
        int chunk = tid * 4 + i;
        int r  = chunk >> 4;
        int c8 = (chunk & 15) << 3;
        uint4 av = zz, bv = zz;
        if (r < SUP) {
            av = *(const uint4*)(cfb + r * LATC + c8);
            bv = *(const uint4*)(tb  + r * LATC + c8);
        }
        *(uint4*)&As[r][c8] = av;
        *(uint4*)&Bt[r][c8] = bv;
    }
    __syncthreads();

    int mt = wave & 3;                       // M tile 0..3
    int nt = wave >> 2;                      // N tiles nt, nt+2
    int rowA = mt * 16 + (lane & 15);
    int rowB0 = nt * 16 + (lane & 15);
    int rowB1 = (nt + 2) * 16 + (lane & 15);
    int ha = (lane >> 4) << 3;               // A K-offset of this half-wave
    int hb = (lane >> 4) << 4;               // unused; B fragments also use ha
    v8f acc0 = {}, acc1 = {};
#pragma unroll
    for (int k0 = 0; k0 < 128; k0 += 32) {
        FragH a, b0, b1;
        a.q[0]  = *(const uint4*)&As[rowA][k0 + ha];
        a.q[1]  = *(const uint4*)&As[rowA][k0 + 16 + ha];
        b0.q[0] = *(const uint4*)&Bt[rowB0][k0 + ha];
        b0.q[1] = *(const uint4*)&Bt[rowB0][k0 + 16 + ha];
        b1.q[0] = *(const uint4*)&Bt[rowB1][k0 + ha];
        b1.q[1] = *(const uint4*)&Bt[rowB1][k0 + 16 + ha];
        acc0 = __builtin_amdgcn_wmma_f32_16x16x32_f16(false, a.v, false, b0.v, (short)0, acc0, false, false);
        acc1 = __builtin_amdgcn_wmma_f32_16x16x32_f16(false, a.v, false, b1.v, (short)0, acc1, false, false);
    }
    half_t* out = CV + (size_t)bid * KP_CV;
#pragma unroll
    for (int v = 0; v < 8; ++v) {
        int m  = mt * 16 + v + ((lane >> 4) << 3);
        int c0 = nt * 16 + (lane & 15);
        int c1 = (nt + 2) * 16 + (lane & 15);
        if (m < SUP && c0 < SUP) out[m * SUP + c0] = (half_t)acc0[v];
        if (m < SUP && c1 < SUP) out[m * SUP + c1] = (half_t)acc1[v];
    }
}

// ---------------------------------------------------------------- generic WMMA GEMM, 64x64 tile, BK=32
// C[rowOut, colOff+cn] = act(A[M,Kp] @ B[Kp,N] + bias) (+ te). M,N mult of 64; Kp mult of 32.
// B is staged transposed (N x K) so fragment reads are contiguous b128.
__global__ __launch_bounds__(256)
void gemm_wmma(const half_t* __restrict__ A, const half_t* __restrict__ B,
               const float* __restrict__ bias, half_t* __restrict__ C,
               int M, int N, int Kp, int ldc, int colOff, int act, int xmode,
               const float* __restrict__ te) {
    __shared__ __align__(16) half_t As[64][40];    // 64 x 32, stride 80B
    __shared__ __align__(16) half_t Bt[64][40];    // B^T: 64 cols x 32 k
    int bm = blockIdx.x, bn = blockIdx.y;
    int tid = threadIdx.x;
    int lane = tid & 31, wave = tid >> 5;
    int mt = wave & 3;
    int nt = wave >> 2;                      // tiles nt and nt+2
    int rowA = mt * 16 + (lane & 15);
    int rowB0 = nt * 16 + (lane & 15);
    int rowB1 = (nt + 2) * 16 + (lane & 15);
    int ha = (lane >> 4) << 3;
    int hb = (lane >> 4) << 4;

    // staging indices
    int ar  = tid >> 2;                      // A: 64 rows, 4 chunks/row
    int ac8 = (tid & 3) << 3;
    int bk  = tid >> 3;                      // B: 32 k-rows, 8 chunks/row
    int bn8 = (tid & 7) << 3;

    v8f acc0 = {}, acc1 = {};
    for (int k0 = 0; k0 < Kp; k0 += 32) {
        uint4 av = *(const uint4*)(A + (size_t)(bm * 64 + ar) * Kp + k0 + ac8);
        uint4 bv = *(const uint4*)(B + (size_t)(k0 + bk) * N + bn * 64 + bn8);
        *(uint4*)&As[ar][ac8] = av;
        const half_t* bh = (const half_t*)&bv;
#pragma unroll
        for (int j = 0; j < 8; ++j) Bt[bn8 + j][bk] = bh[j];   // transpose into LDS
        if (k0 + 32 < Kp) {                  // global_prefetch_b8 of next K tile
            __builtin_prefetch(A + (size_t)(bm * 64 + ar) * Kp + k0 + 32 + ac8, 0, 3);
            __builtin_prefetch(B + (size_t)(k0 + 32 + bk) * N + bn * 64 + bn8, 0, 3);
        }
        __syncthreads();
        FragH a, b0, b1;
        a.q[0]  = *(const uint4*)&As[rowA][ha];
        a.q[1]  = *(const uint4*)&As[rowA][16 + ha];
        b0.q[0] = *(const uint4*)&Bt[rowB0][ha];
        b0.q[1] = *(const uint4*)&Bt[rowB0][16 + ha];
        b1.q[0] = *(const uint4*)&Bt[rowB1][ha];
        b1.q[1] = *(const uint4*)&Bt[rowB1][16 + ha];
        acc0 = __builtin_amdgcn_wmma_f32_16x16x32_f16(false, a.v, false, b0.v, (short)0, acc0, false, false);
        acc1 = __builtin_amdgcn_wmma_f32_16x16x32_f16(false, a.v, false, b1.v, (short)0, acc1, false, false);
        __syncthreads();
    }
#pragma unroll
    for (int v = 0; v < 8; ++v) {
        int r = bm * 64 + mt * 16 + v + ((lane >> 4) << 3);
#pragma unroll
        for (int p = 0; p < 2; ++p) {
            int cn = bn * 64 + (nt + 2 * p) * 16 + (lane & 15);
            float val = (p ? acc1[v] : acc0[v]) + bias[cn];
            if (act) val = gelu_exact(val);
            int rowOut = r;
            if (xmode) {                     // A rows t-major -> X rows n-major, add time emb
                int t = r >> 7, n = r & (NPTS - 1);
                rowOut = n * T_FRAMES + t;
                val += te[t * D_INF + colOff + cn];
            }
            C[(size_t)rowOut * ldc + colOff + cn] = (half_t)val;
        }
    }
}

// ---------------------------------------------------------------- init coords/vis/conf
__global__ void init_state(const float* __restrict__ queries, float* __restrict__ coords,
                           float* __restrict__ vis, float* __restrict__ conf) {
    int tid = blockIdx.x * blockDim.x + threadIdx.x;
    if (tid >= T_FRAMES * NPTS) return;
    int n = tid & (NPTS - 1);
    coords[tid * 2 + 0] = queries[n * 3 + 1] * 0.25f;
    coords[tid * 2 + 1] = queries[n * 3 + 2] * 0.25f;
    vis[tid] = 0.f;
    conf[tid] = 0.f;
}

// ---------------------------------------------------------------- vis/conf/posenc columns of X (+te)
__global__ void build_pe(const float* __restrict__ coords, const float* __restrict__ vis,
                         const float* __restrict__ conf, const float* __restrict__ te,
                         half_t* __restrict__ X) {
    int tid = blockIdx.x * blockDim.x + threadIdx.x;
    if (tid >= T_FRAMES * NPTS) return;
    int t = tid >> 7, n = tid & (NPTS - 1);
    int row = n * T_FRAMES + t;
    float rfx = 0, rfy = 0, rbx = 0, rby = 0;
    if (t < T_FRAMES - 1) {
        rfx = (coords[tid * 2 + 0] - coords[((t + 1) * NPTS + n) * 2 + 0]) / 128.f;
        rfy = (coords[tid * 2 + 1] - coords[((t + 1) * NPTS + n) * 2 + 1]) / 96.f;
    }
    if (t > 0) {
        rbx = (coords[tid * 2 + 0] - coords[((t - 1) * NPTS + n) * 2 + 0]) / 128.f;
        rby = (coords[tid * 2 + 1] - coords[((t - 1) * NPTS + n) * 2 + 1]) / 96.f;
    }
    float x4[4] = {rfx, rfy, rbx, rby};
    const float* ter = te + t * D_INF;
    half_t* Xr = X + (size_t)row * KP_X;
    Xr[0] = (half_t)(vis[tid] + ter[0]);
    Xr[1] = (half_t)(conf[tid] + ter[1]);
    const int base = 2 + 1024;               // 1026
#pragma unroll
    for (int k = 0; k < 4; ++k) Xr[base + k] = (half_t)(x4[k] + ter[base + k]);
#pragma unroll
    for (int d = 0; d < 10; ++d)
#pragma unroll
        for (int k = 0; k < 4; ++k) {
            float xb = x4[k] * (float)(1 << d);
            int i1 = base + 4 + d * 4 + k;        // sin
            int i2 = base + 44 + d * 4 + k;       // cos (= sin(x+pi/2))
            Xr[i1] = (half_t)(sinf(xb) + ter[i1]);
            Xr[i2] = (half_t)(sinf(xb + 1.57079632679f) + ter[i2]);
        }
}

// ---------------------------------------------------------------- UP2 (384->4) + state update
__global__ __launch_bounds__(128)
void up2_update(const half_t* __restrict__ Hb, const float* __restrict__ w2,
                const float* __restrict__ b2, float* __restrict__ coords,
                float* __restrict__ vis, float* __restrict__ conf) {
    __shared__ float red[128][4];
    int row = blockIdx.x;                    // n*T+t
    int n = row >> 4, t = row & 15;
    int tid = threadIdx.x;
    float s0 = 0, s1 = 0, s2 = 0, s3 = 0;
    const half_t* hr = Hb + (size_t)row * 384;
    for (int k = tid; k < 384; k += 128) {
        float h = (float)hr[k];
        const float* wk = w2 + k * 4;
        s0 += h * wk[0]; s1 += h * wk[1]; s2 += h * wk[2]; s3 += h * wk[3];
    }
    red[tid][0] = s0; red[tid][1] = s1; red[tid][2] = s2; red[tid][3] = s3;
    __syncthreads();
    for (int st = 64; st > 0; st >>= 1) {
        if (tid < st)
            for (int j = 0; j < 4; ++j) red[tid][j] += red[tid + st][j];
        __syncthreads();
    }
    if (tid == 0) {
        int tn = t * NPTS + n;
        coords[tn * 2 + 0] += red[0][0] + b2[0];
        coords[tn * 2 + 1] += red[0][1] + b2[1];
        vis[tn]  += red[0][2] + b2[2];
        conf[tn] += red[0][3] + b2[3];
    }
}

// ---------------------------------------------------------------- outputs
__global__ void finalize(const float* __restrict__ coords, const float* __restrict__ vis,
                         const float* __restrict__ conf, float* __restrict__ out) {
    int tid = blockIdx.x * blockDim.x + threadIdx.x;
    if (tid >= T_FRAMES * NPTS) return;
    out[tid * 2 + 0] = coords[tid * 2 + 0] * 4.f;
    out[tid * 2 + 1] = coords[tid * 2 + 1] * 4.f;
    out[4096 + tid] = 1.f / (1.f + expf(-vis[tid]));
    out[6144 + tid] = 1.f / (1.f + expf(-conf[tid]));
}

// ================================================================ host
static inline size_t a256(size_t x) { return (x + 255) & ~(size_t)255; }

extern "C" void kernel_launch(void* const* d_in, const int* in_sizes, int n_in,
                              void* d_out, int out_size, void* d_ws, size_t ws_size,
                              hipStream_t stream) {
    const float* video   = (const float*)d_in[0];
    const float* queries = (const float*)d_in[1];
    const float* fnet_w  = (const float*)d_in[2];
    const float* fnet_b  = (const float*)d_in[3];
    const float* fc1_w   = (const float*)d_in[4];
    const float* fc1_b   = (const float*)d_in[5];
    const float* fc2_w   = (const float*)d_in[6];
    const float* fc2_b   = (const float*)d_in[7];
    const float* up_w1   = (const float*)d_in[8];
    const float* up_b1   = (const float*)d_in[9];
    const float* up_w2   = (const float*)d_in[10];
    const float* up_b2   = (const float*)d_in[11];
    const float* temb    = (const float*)d_in[12];
    const int ITERS = 4;                     // reference setup: iters == 4

    const int Hl[4] = {96, 48, 24, 12};
    const int Wl[4] = {128, 64, 32, 16};
    size_t lvlOff[4];
    {
        size_t o = 0;
        for (int i = 0; i < 4; ++i) { lvlOff[i] = o; o += (size_t)T_FRAMES * Hl[i] * Wl[i] * LATC; }
    }
    const int TN = T_FRAMES * NPTS;          // 2048

    char* ws = (char*)d_ws;
    size_t off = 0;
    auto alloc = [&](size_t bytes) -> void* { void* p = ws + off; off = a256(off + bytes); return p; };

    size_t pyrElems = lvlOff[3] + (size_t)T_FRAMES * Hl[3] * Wl[3] * LATC;
    half_t* PYR  = (half_t*)alloc(pyrElems * 2);
    half_t* TFS  = (half_t*)alloc((size_t)4 * NPTS * SUP * LATC * 2);
    half_t* CF   = (half_t*)alloc((size_t)TN * SUP * LATC * 2);
    size_t cvBytes = (size_t)TN * KP_CV * 2;
    half_t* CV   = (half_t*)alloc(cvBytes);
    half_t* E1   = (half_t*)alloc((size_t)TN * 384 * 2);
    size_t xBytes = (size_t)TN * KP_X * 2;
    half_t* X    = (half_t*)alloc(xBytes);
    half_t* HB   = (half_t*)alloc((size_t)TN * 384 * 2);
    half_t* W1H  = (half_t*)alloc((size_t)KP_CV * 384 * 2);
    half_t* W2H  = (half_t*)alloc((size_t)384 * 256 * 2);
    half_t* U1H  = (half_t*)alloc((size_t)KP_X * 384 * 2);
    float*  TE   = (float*)alloc((size_t)T_FRAMES * D_INF * 4);
    float*  CRD  = (float*)alloc((size_t)TN * 2 * 4);
    float*  VIS  = (float*)alloc((size_t)TN * 4);
    float*  CONF = (float*)alloc((size_t)TN * 4);

    // zero pad columns of CV and X (non-pad entries are overwritten every iteration)
    hipMemsetAsync(CV, 0, cvBytes, stream);
    hipMemsetAsync(X,  0, xBytes,  stream);

    // weight conversions (f32 -> f16, K-padded)
    { int n = KP_CV * 384; cvt_pad_f16<<<(n + 255) / 256, 256, 0, stream>>>(fc1_w, W1H, 2401, KP_CV, 384); }
    { int n = 384 * 256;   cvt_pad_f16<<<(n + 255) / 256, 256, 0, stream>>>(fc2_w, W2H, 384, 384, 256); }
    { int n = KP_X * 384;  cvt_pad_f16<<<(n + 255) / 256, 256, 0, stream>>>(up_w1, U1H, D_INF, KP_X, 384); }
    { int n = T_FRAMES * D_INF; time_interp<<<(n + 255) / 256, 256, 0, stream>>>(temb, TE); }

    // features + pyramid + template features
    fnet_conv<<<T_FRAMES * H0 * W0, 128, 0, stream>>>(video, fnet_w, fnet_b, PYR);
    for (int l = 1; l < 4; ++l)
        pool2<<<T_FRAMES * Hl[l] * Wl[l], 128, 0, stream>>>(PYR + lvlOff[l - 1], PYR + lvlOff[l],
                                                            Hl[l - 1], Wl[l - 1]);
    sample_tfs<<<4 * NPTS * SUP, 128, 0, stream>>>(PYR, queries, TFS);
    init_state<<<(TN + 255) / 256, 256, 0, stream>>>(queries, CRD, VIS, CONF);

    for (int it = 0; it < ITERS; ++it) {
        for (int lvl = 0; lvl < 4; ++lvl) {
            sample_cf<<<TN * SUP, 128, 0, stream>>>(PYR + lvlOff[lvl], CRD, CF, lvl, Hl[lvl], Wl[lvl]);
            cv_wmma<<<TN, 256, 0, stream>>>(CF, TFS, CV, lvl);
            // FC1: [2048,2432] @ [2432,384] + b, gelu -> E1
            gemm_wmma<<<dim3(TN / 64, 384 / 64), 256, 0, stream>>>(
                CV, W1H, fc1_b, E1, TN, 384, KP_CV, 384, 0, 1, 0, nullptr);
            // FC2: [2048,384] @ [384,256] + b -> X[:, 2+lvl*256 : ...] (row remap + time emb)
            gemm_wmma<<<dim3(TN / 64, 256 / 64), 256, 0, stream>>>(
                E1, W2H, fc2_b, X, TN, 256, 384, KP_X, 2 + lvl * 256, 0, 1, TE);
        }
        build_pe<<<(TN + 255) / 256, 256, 0, stream>>>(CRD, VIS, CONF, TE, X);
        // UP1: [2048,1120] @ [1120,384] + b, gelu -> HB
        gemm_wmma<<<dim3(TN / 64, 384 / 64), 256, 0, stream>>>(
            X, U1H, up_b1, HB, TN, 384, KP_X, 384, 0, 1, 0, nullptr);
        up2_update<<<TN, 128, 0, stream>>>(HB, up_w2, up_b2, CRD, VIS, CONF);
    }
    finalize<<<(TN + 255) / 256, 256, 0, stream>>>(CRD, VIS, CONF, (float*)d_out);
}
